// CustomGraphSAGE_18287970746599
// MI455X (gfx1250) — matrix operand order repaired and checked by reading, not compiled, not run on hardware
//
#include <hip/hip_runtime.h>

#define D 64

typedef __attribute__((ext_vector_type(16))) _Float16 v16h;
typedef __attribute__((ext_vector_type(8)))  float    v8f;

// ---------------------------------------------------------------------------
// utility: zero a float region
// ---------------------------------------------------------------------------
__global__ void zero_f32(float* __restrict__ p, int n) {
    int i = blockIdx.x * blockDim.x + threadIdx.x;
    if (i < n) p[i] = 0.0f;
}

// ---------------------------------------------------------------------------
// degree counts: counts[dst[e]] += 1.0f   (hardware f32 atomics in L2)
// ---------------------------------------------------------------------------
__global__ void count_edges(const long long* __restrict__ dst,
                            float* __restrict__ counts, int E) {
    int e = blockIdx.x * blockDim.x + threadIdx.x;
    if (e < E) unsafeAtomicAdd(&counts[(int)dst[e]], 1.0f);
}

// ---------------------------------------------------------------------------
// in-place: counts[n] -> 1 / max(counts[n], 1)   (keeps divides out of GEMM)
// ---------------------------------------------------------------------------
__global__ void recip_counts(float* __restrict__ counts, int n) {
    int i = blockIdx.x * blockDim.x + threadIdx.x;
    if (i < n) counts[i] = 1.0f / fmaxf(counts[i], 1.0f);
}

// ---------------------------------------------------------------------------
// scatter messages: summed[dst] += x[src] * ew   (wp folded into GEMM scale)
// one thread per (edge, 4-dim quad): 16 consecutive lanes stream one 256B row
// ---------------------------------------------------------------------------
__global__ void scatter_msgs(const float* __restrict__ xin,
                             const long long* __restrict__ src,
                             const long long* __restrict__ dst,
                             const float* __restrict__ ew,
                             float* __restrict__ summed, int E) {
    long long tid = (long long)blockIdx.x * blockDim.x + threadIdx.x;
    int e = (int)(tid >> 4);
    int q = (int)(tid & 15);
    if (e >= E) return;

    int   s = (int)src[e];
    int   d = (int)dst[e];
    float w = ew[e];

    // speculative prefetch of a future edge's source row (global_prefetch_b8)
    int ep = e + 2048;
    if (ep < E) {
        const float* pf = xin + (size_t)((int)src[ep]) * D + q * 4;
        __builtin_prefetch(pf, 0, 1);
    }

    float4 v = *(const float4*)(xin + (size_t)s * D + q * 4);
    float* o = summed + (size_t)d * D + q * 4;
    unsafeAtomicAdd(o + 0, v.x * w);
    unsafeAtomicAdd(o + 1, v.y * w);
    unsafeAtomicAdd(o + 2, v.z * w);
    unsafeAtomicAdd(o + 3, v.w * w);
}

// ---------------------------------------------------------------------------
// WMMA helpers (ISA 7.12.2 fragment layouts, wave32)
// ---------------------------------------------------------------------------
__device__ __forceinline__ v8f wmma_f16(v16h a, v16h b, v8f c) {
    return __builtin_amdgcn_wmma_f32_16x16x32_f16(
        /*neg_a=*/false, a, /*neg_b=*/false, b,
        /*c_mod=*/(short)0, c, /*reuse_a=*/false, /*reuse_b=*/false);
}

// B fragment (32x16, K x N): lane holds N = lane%16 (W row); elements 0..7 cover
// K = kbase + i, 8..15 cover K = kbase + 16 + i, kbase = ktile*32 + (lane/16)*8.  B[k][n] = W[n][k]  (x @ W^T).
__device__ __forceinline__ v16h load_b_frag(const float* __restrict__ W,
                                            int n, int kbase) {
    v16h b;
    const float* p = W + n * D + kbase;
#pragma unroll
    for (int j = 0; j < 16; ++j) b[j] = (_Float16)p[j < 8 ? j : j + 8];
    return b;
}

// A fragment (16x32, M x K): lane holds M = lane%16; halves 0..7 cover
// K = kt*32 + g*8 + i, halves 8..15 cover K = kt*32 + 16 + g*8 + i  (g=lane/16)
__device__ __forceinline__ v16h load_a_frag(const float* __restrict__ X,
                                            int row, int kt, int g, float scale) {
    v16h a;
    const float* p = X + (size_t)row * D + kt * 32 + g * 8;
#pragma unroll
    for (int i = 0; i < 8; ++i) a[i]     = (_Float16)(p[i]      * scale);
#pragma unroll
    for (int i = 0; i < 8; ++i) a[i + 8] = (_Float16)(p[16 + i] * scale);
    return a;
}

// C/D fragment store: VGPR r -> M = r + 8*(lane/16); N = lane%16
template <int DO_RELU>
__device__ __forceinline__ void store_tile(float* __restrict__ out, v8f c,
                                           int tile, int nt, int g, int n16,
                                           float bias, int nrows) {
    int col = nt * 16 + n16;
#pragma unroll
    for (int r = 0; r < 8; ++r) {
        int m = tile * 16 + g * 8 + r;
        if (m < nrows) {
            float v = c[r] + bias;
            if (DO_RELU) v = fmaxf(v, 0.0f);
            out[(size_t)m * D + col] = v;
        }
    }
}

// ---------------------------------------------------------------------------
// fused SAGE layer epilogue:
//   out = x @ Ws^T + bs + (summed * wp * inv_count) @ Wn^T + bn   (+ReLU)
// one wave computes a 16x64 output tile with 16 chained v_wmma ops.
// ---------------------------------------------------------------------------
template <int DO_RELU>
__global__ void __launch_bounds__(256)
sage_gemm(const float* __restrict__ xin, const float* __restrict__ summed,
          const float* __restrict__ invc,           // 1/max(count,1), precomputed
          const float* __restrict__ Ws, const float* __restrict__ bs,
          const float* __restrict__ Wn, const float* __restrict__ bn,
          const float* __restrict__ wp,
          float* __restrict__ out, int nrows) {
    const int lane   = threadIdx.x & 31;
    const int g      = lane >> 4;
    const int n16    = lane & 15;
    const int waveId = (blockIdx.x * blockDim.x + threadIdx.x) >> 5;
    const int nwaves = (gridDim.x * blockDim.x) >> 5;
    const int ntiles = (nrows + 15) >> 4;
    const float wscale = wp[0];

    // hoist weight fragments (f32 -> f16) and per-lane biases out of the loop
    v16h bw_s[2][4], bw_n[2][4];
    float bias[4];
#pragma unroll
    for (int nt = 0; nt < 4; ++nt) {
        int n = nt * 16 + n16;
        bias[nt] = bs[n] + bn[n];
#pragma unroll
        for (int kt = 0; kt < 2; ++kt) {
            int kbase = kt * 32 + g * 8;
            bw_s[kt][nt] = load_b_frag(Ws, n, kbase);
            bw_n[kt][nt] = load_b_frag(Wn, n, kbase);
        }
    }

    for (int tile = waveId; tile < ntiles; tile += nwaves) {
        int rowA = tile * 16 + n16;
        if (rowA >= nrows) rowA = nrows - 1;   // clamp (N%16==0 in practice)

        v8f c0 = {}, c1 = {}, c2 = {}, c3 = {};

        // self term: x @ Ws^T
#pragma unroll
        for (int kt = 0; kt < 2; ++kt) {
            v16h a = load_a_frag(xin, rowA, kt, g, 1.0f);
            c0 = wmma_f16(a, bw_s[kt][0], c0);
            c1 = wmma_f16(a, bw_s[kt][1], c1);
            c2 = wmma_f16(a, bw_s[kt][2], c2);
            c3 = wmma_f16(a, bw_s[kt][3], c3);
        }
        // neighbor term: mean-scaled aggregate @ Wn^T (wp * 1/deg, no divide)
        float inv = wscale * invc[rowA];
#pragma unroll
        for (int kt = 0; kt < 2; ++kt) {
            v16h a = load_a_frag(summed, rowA, kt, g, inv);
            c0 = wmma_f16(a, bw_n[kt][0], c0);
            c1 = wmma_f16(a, bw_n[kt][1], c1);
            c2 = wmma_f16(a, bw_n[kt][2], c2);
            c3 = wmma_f16(a, bw_n[kt][3], c3);
        }

        store_tile<DO_RELU>(out, c0, tile, 0, g, n16, bias[0], nrows);
        store_tile<DO_RELU>(out, c1, tile, 1, g, n16, bias[1], nrows);
        store_tile<DO_RELU>(out, c2, tile, 2, g, n16, bias[2], nrows);
        store_tile<DO_RELU>(out, c3, tile, 3, g, n16, bias[3], nrows);
    }
}

// ---------------------------------------------------------------------------
// launch
// ---------------------------------------------------------------------------
extern "C" void kernel_launch(void* const* d_in, const int* in_sizes, int n_in,
                              void* d_out, int out_size, void* d_ws, size_t ws_size,
                              hipStream_t stream) {
    const float*     x   = (const float*)d_in[0];
    const long long* ei  = (const long long*)d_in[1];   // (2,E) int64
    const float*     ew  = (const float*)d_in[2];
    const float*     Ws1 = (const float*)d_in[3];
    const float*     bs1 = (const float*)d_in[4];
    const float*     Wn1 = (const float*)d_in[5];
    const float*     bn1 = (const float*)d_in[6];
    const float*     wp1 = (const float*)d_in[7];
    const float*     Ws2 = (const float*)d_in[8];
    const float*     bs2 = (const float*)d_in[9];
    const float*     Wn2 = (const float*)d_in[10];
    const float*     bn2 = (const float*)d_in[11];
    const float*     wp2 = (const float*)d_in[12];

    const int N = in_sizes[0] / D;
    const int E = in_sizes[2];
    const long long* src = ei;
    const long long* dst = ei + E;

    // workspace layout: invc[N] | summed[N*D] | h[N*D]
    size_t countsBytes = ((size_t)N * 4 + 255) & ~(size_t)255;
    size_t summedBytes = ((size_t)N * D * 4 + 255) & ~(size_t)255;
    float* invc   = (float*)d_ws;
    float* summed = (float*)((char*)d_ws + countsBytes);
    float* h      = (float*)((char*)d_ws + countsBytes + summedBytes);
    float* out    = (float*)d_out;

    const int T = 256;
    const int nND = N * D;

    // --- shared degree counts -> reciprocals (same for both layers) ---
    zero_f32<<<(N + T - 1) / T, T, 0, stream>>>(invc, N);
    count_edges<<<(E + T - 1) / T, T, 0, stream>>>(dst, invc, E);
    recip_counts<<<(N + T - 1) / T, T, 0, stream>>>(invc, N);

    // --- layer 1 ---
    zero_f32<<<(nND + T - 1) / T, T, 0, stream>>>(summed, nND);
    {
        long long nt = (long long)E * 16;
        scatter_msgs<<<(unsigned)((nt + T - 1) / T), T, 0, stream>>>(x, src, dst, ew, summed, E);
    }
    sage_gemm<1><<<256, T, 0, stream>>>(x, summed, invc, Ws1, bs1, Wn1, bn1, wp1, h, N);

    // --- layer 2 ---
    zero_f32<<<(nND + T - 1) / T, T, 0, stream>>>(summed, nND);
    {
        long long nt = (long long)E * 16;
        scatter_msgs<<<(unsigned)((nt + T - 1) / T), T, 0, stream>>>(h, src, dst, ew, summed, E);
    }
    sage_gemm<0><<<256, T, 0, stream>>>(h, summed, invc, Ws2, bs2, Wn2, bn2, wp2, out, N);
}
